// TPPInteraction_70944269796235
// MI455X (gfx1250) — compile-verified, matrix operand order checked by reading, not run on hardware
//
#include <hip/hip_runtime.h>
#include <math.h>

// ---------------------------------------------------------------------------
// Types for CDNA5 WMMA (wave32): v_wmma_f32_16x16x32_f16
// ---------------------------------------------------------------------------
typedef _Float16 h8_t  __attribute__((ext_vector_type(8)));
typedef _Float16 h16_t __attribute__((ext_vector_type(16)));
typedef float    f8_t  __attribute__((ext_vector_type(8)));

#define DS_ 128
#define DV_ 16

__device__ __forceinline__ float sigmoid_f(float x) { return 1.0f / (1.0f + __expf(-x)); }
__device__ __forceinline__ float silu_f(float x)    { return x * sigmoid_f(x); }

// A-matrix fragment (16x32 f16, row-major rows in LDS, row stride = strideK halves).
// CDNA5 16-bit A layout: lane L holds row (L&15); lanes 0-15 cover K = 0..7,16..23
// of the k-step, lanes 16-31 cover K = 8..15,24..31.
__device__ __forceinline__ h16_t load_a_frag(const _Float16* rows, int strideK, int kstep, int lane) {
    const int row = lane & 15;
    const int kb  = kstep * 32 + ((lane >> 4) << 3);
    const _Float16* p = rows + row * strideK + kb;
    h8_t lo = *(const h8_t*)(p);
    h8_t hi = *(const h8_t*)(p + 16);
    h16_t r;
#pragma unroll
    for (int i = 0; i < 8; ++i) { r[i] = lo[i]; r[8 + i] = hi[i]; }
    return r;
}

// B fragments are pre-packed so each lane's 16 halves are contiguous:
// layout [frag][lane][16], frag stride 512 halves.
__device__ __forceinline__ h16_t load_b_frag(const _Float16* packed, int frag, int lane) {
    return *(const h16_t*)(packed + ((size_t)frag << 9) + (lane << 4));
}

__device__ __forceinline__ f8_t wmma_f16(h16_t a, h16_t b, f8_t c) {
    // (neg_a, A, neg_b, B, c_mod, C, reuse_a, reuse_b)
    return __builtin_amdgcn_wmma_f32_16x16x32_f16(false, a, false, b, (short)0, c, false, false);
}

// ---------------------------------------------------------------------------
// Weight repack: row-major f32 [K][N] -> fragment-ready f16 [nt][ks][lane][16]
// matching the B-operand striping assumed by load_b_frag.
// ---------------------------------------------------------------------------
__global__ void pack_w(const float* __restrict__ W, _Float16* __restrict__ out, int K, int Nw) {
    const int ks = K / 32;
    const int total = K * Nw;
    for (int idx = blockIdx.x * blockDim.x + threadIdx.x; idx < total; idx += gridDim.x * blockDim.x) {
        int t   = idx / (ks * 512);
        int rem = idx - t * (ks * 512);
        int k   = rem / 512;
        int l   = rem - k * 512;
        int lane = l >> 4;
        int i    = l & 15;
        int col  = t * 16 + (lane & 15);
        int kb   = k * 32 + ((lane >> 4) << 3);
        int kk   = (i < 8) ? (kb + i) : (kb + 16 + (i - 8));
        out[idx] = (_Float16)W[(size_t)kk * Nw + col];
    }
}

// ---------------------------------------------------------------------------
// Edge kernel: TPP message + attention + scatter-sum. 2 waves/block, each wave
// owns one 16-edge WMMA tile.
// ---------------------------------------------------------------------------
__global__ __launch_bounds__(64) void edge_kernel(
    const float* __restrict__ node_s, const float* __restrict__ node_V,
    const float* __restrict__ frames, const int* __restrict__ eidx,
    const float* __restrict__ Wvd,  const float* __restrict__ Wvu,
    const float* __restrict__ bs,   const float* __restrict__ bg,
    const float* __restrict__ Watt, const float* __restrict__ batt,
    const _Float16* __restrict__ Wsm_p, const _Float16* __restrict__ Wgm_p,
    float* __restrict__ agg_s, float* __restrict__ agg_V, int E)
{
    __shared__ _Float16 A[2][16][320];       // [s_src|s_dst|norms|proj] f16 A-panel
    __shared__ float    VeS[2][16][32][3];   // gathered V_e
    __shared__ float    VdS[2][16][16][3];   // down-projected vectors
    __shared__ float    soS[2][16][128];     // silu(so) f32
    __shared__ _Float16 soH[2][16][128];     // silu(so) f16 (gate GEMM A)
    __shared__ int      dstS[2][16];
    __shared__ float    attS[2][16];
    __shared__ float    red[2][16][2];

    const int lane = threadIdx.x & 31;
    const int w    = threadIdx.x >> 5;
    const int tile = blockIdx.x * 2 + w;
    const int e0   = tile * 16;

    // ---- phase 1: gather scalars + vectors (2 lanes per edge) ----
    {
        const int m = lane >> 1, p = lane & 1;
        const int e   = e0 + m;
        const int src = eidx[e];
        const int dst = eidx[E + e];
        if (p == 0) dstS[w][m] = dst;
        const int node = p ? dst : src;
        const float* srow = node_s + (size_t)node * DS_;
#pragma unroll
        for (int c = 0; c < DS_; c += 4) {
            float4 v = *(const float4*)(srow + c);
            A[w][m][p * DS_ + c + 0] = (_Float16)v.x;
            A[w][m][p * DS_ + c + 1] = (_Float16)v.y;
            A[w][m][p * DS_ + c + 2] = (_Float16)v.z;
            A[w][m][p * DS_ + c + 3] = (_Float16)v.w;
        }
        const float* vrow = node_V + (size_t)node * (DV_ * 3);
#pragma unroll
        for (int vv = 0; vv < DV_; ++vv) {
            VeS[w][m][p * DV_ + vv][0] = vrow[vv * 3 + 0];
            VeS[w][m][p * DV_ + vv][1] = vrow[vv * 3 + 1];
            VeS[w][m][p * DV_ + vv][2] = vrow[vv * 3 + 2];
        }
    }
    __syncthreads();

    // ---- phase 2: Vd = V_e @ Wvd, norms, frame scalarization ----
    {
        const int m = lane >> 1, p = lane & 1;
        const int dst = dstS[w][m];
        float fr[9];
#pragma unroll
        for (int i = 0; i < 9; ++i) fr[i] = frames[(size_t)dst * 9 + i];
        for (int hh = 0; hh < 8; ++hh) {
            const int h = p * 8 + hh;
            float d0 = 0.f, d1 = 0.f, d2 = 0.f;
            for (int v = 0; v < 32; ++v) {
                const float wv = Wvd[v * DV_ + h];
                d0 += VeS[w][m][v][0] * wv;
                d1 += VeS[w][m][v][1] * wv;
                d2 += VeS[w][m][v][2] * wv;
            }
            VdS[w][m][h][0] = d0; VdS[w][m][h][1] = d1; VdS[w][m][h][2] = d2;
            A[w][m][256 + h] = (_Float16)sqrtf(d0 * d0 + d1 * d1 + d2 * d2 + 1e-8f);
#pragma unroll
            for (int c = 0; c < 3; ++c)
                A[w][m][272 + h * 3 + c] =
                    (_Float16)(d0 * fr[c * 3 + 0] + d1 * fr[c * 3 + 1] + d2 * fr[c * 3 + 2]);
        }
    }
    __syncthreads();

    // ---- phase 3: so[16,128] = A[16,320] @ Ws_m  (WMMA, K=320 -> 10 steps) ----
    const f8_t fz = {0.f, 0.f, 0.f, 0.f, 0.f, 0.f, 0.f, 0.f};
    f8_t acc[8];
#pragma unroll
    for (int t = 0; t < 8; ++t) acc[t] = fz;
    for (int k = 0; k < 10; ++k) {
        h16_t av = load_a_frag(&A[w][0][0], 320, k, lane);
#pragma unroll
        for (int t = 0; t < 8; ++t) {
            h16_t bv = load_b_frag(Wsm_p, t * 10 + k, lane);
            acc[t] = wmma_f16(av, bv, acc[t]);
        }
    }
    {   // epilogue: +bias, silu. C/D layout: row = r + 8*(lane>>4), col = t*16 + (lane&15)
        const int half = lane >> 4, col0 = lane & 15;
#pragma unroll
        for (int t = 0; t < 8; ++t) {
            const int col = t * 16 + col0;
            const float b = bs[col];
#pragma unroll
            for (int r = 0; r < 8; ++r) {
                const int m = r + 8 * half;
                const float so = silu_f(acc[t][r] + b);
                soS[w][m][col] = so;
                soH[w][m][col] = (_Float16)so;
            }
        }
    }
    __syncthreads();

    // ---- phase 4: attention att = sigmoid(so . Watt + batt) ----
    {
        const int m = lane >> 1, p = lane & 1;
        float part = 0.f;
        for (int c = 0; c < 64; ++c) part += soS[w][m][p * 64 + c] * Watt[p * 64 + c];
        red[w][m][p] = part;
    }
    __syncthreads();
    {
        const int m = lane >> 1, p = lane & 1;
        if (p == 0) attS[w][m] = sigmoid_f(red[w][m][0] + red[w][m][1] + batt[0]);
    }
    __syncthreads();

    // ---- phase 5: gate = sigmoid(so @ Wg_m + bg) (WMMA, K=128 -> 4 steps) ----
    f8_t g = fz;
    for (int k = 0; k < 4; ++k) {
        h16_t av = load_a_frag(&soH[w][0][0], 128, k, lane);
        h16_t bv = load_b_frag(Wgm_p, k, lane);
        g = wmma_f16(av, bv, g);
    }

    // ---- phase 6: Vo = (Vd @ Wvu) * gate * att, scatter-add ----
    {
        const int o = lane & 15, half = lane >> 4;
        float wvu[16];
#pragma unroll
        for (int h = 0; h < 16; ++h) wvu[h] = Wvu[h * DV_ + o];
        const float bgo = bg[o];
#pragma unroll
        for (int r = 0; r < 8; ++r) {
            const int m   = r + 8 * half;
            const float a = attS[w][m];
            const float gg = sigmoid_f(g[r] + bgo) * a;
            const int dst = dstS[w][m];
#pragma unroll
            for (int i = 0; i < 3; ++i) {
                float s = 0.f;
#pragma unroll
                for (int h = 0; h < 16; ++h) s += VdS[w][m][h][i] * wvu[h];
                atomicAdd(&agg_V[(size_t)dst * 48 + o * 3 + i], s * gg);
            }
        }
    }
    // ---- phase 7: ms = so*att, scatter-add ----
    for (int idx = lane; idx < 16 * DS_; idx += 32) {
        const int m = idx >> 7, c = idx & 127;
        atomicAdd(&agg_s[(size_t)dstS[w][m] * DS_ + c], soS[w][m][c] * attS[w][m]);
    }
}

// ---------------------------------------------------------------------------
// Node kernel: FFN TPP-1 + TPP-2 + residual + TPPNorm. One wave per 16 nodes.
// ---------------------------------------------------------------------------
__global__ __launch_bounds__(32) void node_kernel(
    const float* __restrict__ node_s, const float* __restrict__ node_V,
    const float* __restrict__ frames,
    const float* __restrict__ agg_s,  const float* __restrict__ agg_V,
    const float* __restrict__ bs1, const float* __restrict__ bg1,
    const float* __restrict__ bs2, const float* __restrict__ bg2,
    const float* __restrict__ Wvd1, const float* __restrict__ Wvu1,
    const float* __restrict__ Wvd2, const float* __restrict__ Wvu2,
    const _Float16* __restrict__ Ws1_p, const _Float16* __restrict__ Wg1_p,
    const _Float16* __restrict__ Ws2_p, const _Float16* __restrict__ Wg2_p,
    const float* __restrict__ gamma, const float* __restrict__ beta,
    float* __restrict__ out_s, float* __restrict__ out_V)
{
    __shared__ _Float16 A1[16][384];         // [hs0|norms1|proj1]
    __shared__ _Float16 A2[16][576];         // [silu(so1)|norms2|proj2]
    __shared__ float    Vd1S[16][32][3];
    __shared__ float    hV1S[16][32][3];     // stages hV0, later holds hV1
    __shared__ float    Vd2S[16][16][3];
    __shared__ float    so2S[16][128];
    __shared__ _Float16 so2H[16][128];
    __shared__ float    VoutS[16][16][3];
    __shared__ float    red[16][2][2];
    __shared__ float    vred[16][2];
    __shared__ float    muS[16], rsS[16], vnS[16];

    const int lane = threadIdx.x & 31;
    const int n0   = blockIdx.x * 16;
    const f8_t fz  = {0.f, 0.f, 0.f, 0.f, 0.f, 0.f, 0.f, 0.f};

    // ---- phase 1: stage hs0 = [agg_s|node_s], hV0 = [agg_V;node_V] ----
    {
        const int m = lane >> 1, p = lane & 1;
        const int n = n0 + m;
        const float* s0 = p ? (node_s + (size_t)n * DS_) : (agg_s + (size_t)n * DS_);
#pragma unroll
        for (int c = 0; c < DS_; c += 4) {
            float4 v = *(const float4*)(s0 + c);
            A1[m][p * DS_ + c + 0] = (_Float16)v.x;
            A1[m][p * DS_ + c + 1] = (_Float16)v.y;
            A1[m][p * DS_ + c + 2] = (_Float16)v.z;
            A1[m][p * DS_ + c + 3] = (_Float16)v.w;
        }
        const float* v0 = p ? (node_V + (size_t)n * 48) : (agg_V + (size_t)n * 48);
#pragma unroll
        for (int vv = 0; vv < 16; ++vv) {
            hV1S[m][p * 16 + vv][0] = v0[vv * 3 + 0];
            hV1S[m][p * 16 + vv][1] = v0[vv * 3 + 1];
            hV1S[m][p * 16 + vv][2] = v0[vv * 3 + 2];
        }
    }
    __syncthreads();

    // ---- phase 2: Vd1 = hV0 @ Wvd1 (32->32), norms1, proj1 ----
    {
        const int m = lane >> 1, p = lane & 1;
        const int n = n0 + m;
        float fr[9];
#pragma unroll
        for (int i = 0; i < 9; ++i) fr[i] = frames[(size_t)n * 9 + i];
        for (int hh = 0; hh < 16; ++hh) {
            const int h = p * 16 + hh;
            float d0 = 0.f, d1 = 0.f, d2 = 0.f;
            for (int v = 0; v < 32; ++v) {
                const float wv = Wvd1[v * 32 + h];
                d0 += hV1S[m][v][0] * wv;
                d1 += hV1S[m][v][1] * wv;
                d2 += hV1S[m][v][2] * wv;
            }
            Vd1S[m][h][0] = d0; Vd1S[m][h][1] = d1; Vd1S[m][h][2] = d2;
            A1[m][256 + h] = (_Float16)sqrtf(d0 * d0 + d1 * d1 + d2 * d2 + 1e-8f);
#pragma unroll
            for (int c = 0; c < 3; ++c)
                A1[m][288 + h * 3 + c] =
                    (_Float16)(d0 * fr[c * 3 + 0] + d1 * fr[c * 3 + 1] + d2 * fr[c * 3 + 2]);
        }
    }
    __syncthreads();

    // ---- phase 3: so1[16,512] = A1[16,384] @ Ws1, n-chunked (4 x 8 tiles) ----
    for (int chunk = 0; chunk < 4; ++chunk) {
        f8_t acc[8];
#pragma unroll
        for (int t = 0; t < 8; ++t) acc[t] = fz;
        for (int k = 0; k < 12; ++k) {
            h16_t av = load_a_frag(&A1[0][0], 384, k, lane);
#pragma unroll
            for (int t = 0; t < 8; ++t) {
                h16_t bv = load_b_frag(Ws1_p, (chunk * 8 + t) * 12 + k, lane);
                acc[t] = wmma_f16(av, bv, acc[t]);
            }
        }
        const int half = lane >> 4, col0 = lane & 15;
#pragma unroll
        for (int t = 0; t < 8; ++t) {
            const int col = chunk * 128 + t * 16 + col0;
            const float b = bs1[col];
#pragma unroll
            for (int r = 0; r < 8; ++r)
                A2[r + 8 * half][col] = (_Float16)silu_f(acc[t][r] + b);
        }
    }
    __syncthreads();

    // ---- phase 4: gate1 = sigmoid(so1 @ Wg1 + bg1) (K=512 region of A2) ----
    f8_t g1[2] = {fz, fz};
    for (int k = 0; k < 16; ++k) {
        h16_t av = load_a_frag(&A2[0][0], 576, k, lane);
#pragma unroll
        for (int t = 0; t < 2; ++t) {
            h16_t bv = load_b_frag(Wg1_p, t * 16 + k, lane);
            g1[t] = wmma_f16(av, bv, g1[t]);
        }
    }
    // hV1 = (Vd1 @ Wvu1) * gate1
    {
        const int half = lane >> 4, col0 = lane & 15;
        for (int t = 0; t < 2; ++t) {
            const int o = t * 16 + col0;
            float wv[32];
#pragma unroll
            for (int h = 0; h < 32; ++h) wv[h] = Wvu1[h * 32 + o];
            const float bo = bg1[o];
#pragma unroll
            for (int r = 0; r < 8; ++r) {
                const int m = r + 8 * half;
                const float gg = sigmoid_f(g1[t][r] + bo);
#pragma unroll
                for (int i = 0; i < 3; ++i) {
                    float s = 0.f;
#pragma unroll
                    for (int h = 0; h < 32; ++h) s += Vd1S[m][h][i] * wv[h];
                    hV1S[m][o][i] = s * gg;
                }
            }
        }
    }
    __syncthreads();

    // ---- phase 5: Vd2 = hV1 @ Wvd2 (32->16), norms2, proj2 into A2[:,512:576] ----
    {
        const int m = lane >> 1, p = lane & 1;
        const int n = n0 + m;
        float fr[9];
#pragma unroll
        for (int i = 0; i < 9; ++i) fr[i] = frames[(size_t)n * 9 + i];
        for (int hh = 0; hh < 8; ++hh) {
            const int h = p * 8 + hh;
            float d0 = 0.f, d1 = 0.f, d2 = 0.f;
            for (int v = 0; v < 32; ++v) {
                const float wv = Wvd2[v * DV_ + h];
                d0 += hV1S[m][v][0] * wv;
                d1 += hV1S[m][v][1] * wv;
                d2 += hV1S[m][v][2] * wv;
            }
            Vd2S[m][h][0] = d0; Vd2S[m][h][1] = d1; Vd2S[m][h][2] = d2;
            A2[m][512 + h] = (_Float16)sqrtf(d0 * d0 + d1 * d1 + d2 * d2 + 1e-8f);
#pragma unroll
            for (int c = 0; c < 3; ++c)
                A2[m][528 + h * 3 + c] =
                    (_Float16)(d0 * fr[c * 3 + 0] + d1 * fr[c * 3 + 1] + d2 * fr[c * 3 + 2]);
        }
    }
    __syncthreads();

    // ---- phase 6: so2[16,128] = A2[16,576] @ Ws2 (no activation) ----
    {
        f8_t acc[8];
#pragma unroll
        for (int t = 0; t < 8; ++t) acc[t] = fz;
        for (int k = 0; k < 18; ++k) {
            h16_t av = load_a_frag(&A2[0][0], 576, k, lane);
#pragma unroll
            for (int t = 0; t < 8; ++t) {
                h16_t bv = load_b_frag(Ws2_p, t * 18 + k, lane);
                acc[t] = wmma_f16(av, bv, acc[t]);
            }
        }
        const int half = lane >> 4, col0 = lane & 15;
#pragma unroll
        for (int t = 0; t < 8; ++t) {
            const int col = t * 16 + col0;
            const float b = bs2[col];
#pragma unroll
            for (int r = 0; r < 8; ++r) {
                const int m = r + 8 * half;
                const float so = acc[t][r] + b;
                so2S[m][col] = so;
                so2H[m][col] = (_Float16)so;
            }
        }
    }
    __syncthreads();

    // ---- phase 7: gate2 + hV2, Vout = node_V + hV2 ----
    f8_t g2 = fz;
    for (int k = 0; k < 4; ++k) {
        h16_t av = load_a_frag(&so2H[0][0], 128, k, lane);
        h16_t bv = load_b_frag(Wg2_p, k, lane);
        g2 = wmma_f16(av, bv, g2);
    }
    {
        const int o = lane & 15, half = lane >> 4;
        float wv[16];
#pragma unroll
        for (int h = 0; h < 16; ++h) wv[h] = Wvu2[h * DV_ + o];
        const float bo = bg2[o];
#pragma unroll
        for (int r = 0; r < 8; ++r) {
            const int m  = r + 8 * half;
            const float gg = sigmoid_f(g2[r] + bo);
#pragma unroll
            for (int i = 0; i < 3; ++i) {
                float s = 0.f;
#pragma unroll
                for (int h = 0; h < 16; ++h) s += Vd2S[m][h][i] * wv[h];
                VoutS[m][o][i] = node_V[(size_t)(n0 + m) * 48 + o * 3 + i] + s * gg;
            }
        }
    }
    __syncthreads();

    // ---- phase 8: residual + LayerNorm(s) + vector-norm(V) ----
    {
        const int m = lane >> 1, p = lane & 1;
        const int n = n0 + m;
        const float* srow = node_s + (size_t)n * DS_;
        float sum = 0.f, sq = 0.f;
        for (int c = p * 64; c < p * 64 + 64; ++c) {
            const float s = srow[c] + so2S[m][c];
            sum += s; sq += s * s;
        }
        red[m][p][0] = sum; red[m][p][1] = sq;
        float vp = 0.f;
        const float* vf = &VoutS[m][0][0];
        for (int j = 0; j < 24; ++j) { const float v = vf[p * 24 + j]; vp += v * v; }
        vred[m][p] = vp;
    }
    __syncthreads();
    {
        const int m = lane >> 1, p = lane & 1;
        if (p == 0) {
            const float sum = red[m][0][0] + red[m][1][0];
            const float sq  = red[m][0][1] + red[m][1][1];
            const float mu  = sum * (1.0f / 128.0f);
            const float var = sq * (1.0f / 128.0f) - mu * mu;
            muS[m] = mu;
            rsS[m] = rsqrtf(var + 1e-5f);
            vnS[m] = sqrtf((vred[m][0] + vred[m][1]) * (1.0f / 16.0f) + 1e-8f);
        }
    }
    __syncthreads();
    {
        const int m = lane >> 1, p = lane & 1;
        const int n = n0 + m;
        const float* srow = node_s + (size_t)n * DS_;
        const float mu = muS[m], rs = rsS[m], inv = 1.0f / vnS[m];
        for (int c = p * 64; c < p * 64 + 64; ++c) {
            const float s = srow[c] + so2S[m][c];
            out_s[(size_t)n * DS_ + c] = (s - mu) * rs * gamma[c] + beta[c];
        }
        const float* vf = &VoutS[m][0][0];
        for (int j = 0; j < 24; ++j) {
            const int idx = p * 24 + j;
            out_V[(size_t)n * 48 + idx] = vf[idx] * inv;
        }
    }
}

// ---------------------------------------------------------------------------
// Host launcher
// ---------------------------------------------------------------------------
extern "C" void kernel_launch(void* const* d_in, const int* in_sizes, int n_in,
                              void* d_out, int out_size, void* d_ws, size_t ws_size,
                              hipStream_t stream) {
    const float* node_s = (const float*)d_in[0];
    const float* node_V = (const float*)d_in[1];
    const float* frames = (const float*)d_in[2];
    const int*   eidx   = (const int*)d_in[3];
    const float* Wvd_m  = (const float*)d_in[4];
    const float* Wvu_m  = (const float*)d_in[5];
    const float* Ws_m   = (const float*)d_in[6];
    const float* bs_m   = (const float*)d_in[7];
    // d_in[8] = Wg_m (packed below)
    const float* bg_m   = (const float*)d_in[9];
    const float* Watt   = (const float*)d_in[10];
    const float* batt   = (const float*)d_in[11];
    const float* Wvd1   = (const float*)d_in[12];
    const float* Wvu1   = (const float*)d_in[13];
    const float* Ws1    = (const float*)d_in[14];
    const float* bs1    = (const float*)d_in[15];
    const float* Wg1    = (const float*)d_in[16];
    const float* bg1    = (const float*)d_in[17];
    const float* Wvd2   = (const float*)d_in[18];
    const float* Wvu2   = (const float*)d_in[19];
    const float* Ws2    = (const float*)d_in[20];
    const float* bs2    = (const float*)d_in[21];
    const float* Wg2    = (const float*)d_in[22];
    const float* bg2    = (const float*)d_in[23];
    const float* gamma  = (const float*)d_in[24];
    const float* beta   = (const float*)d_in[25];

    const int N = in_sizes[0] / DS_;
    const int E = in_sizes[3] / 2;

    char* ws = (char*)d_ws;
    size_t off = 0;
    auto alloc = [&](size_t bytes) -> char* {
        off = (off + 255) & ~(size_t)255;
        char* p = ws + off;
        off += bytes;
        return p;
    };
    float*    agg_s = (float*)alloc((size_t)N * DS_ * sizeof(float));
    float*    agg_V = (float*)alloc((size_t)N * 48 * sizeof(float));
    _Float16* Wsm_p = (_Float16*)alloc(320 * 128 * sizeof(_Float16));
    _Float16* Wgm_p = (_Float16*)alloc(128 * 16 * sizeof(_Float16));
    _Float16* Ws1_p = (_Float16*)alloc(384 * 512 * sizeof(_Float16));
    _Float16* Wg1_p = (_Float16*)alloc(512 * 32 * sizeof(_Float16));
    _Float16* Ws2_p = (_Float16*)alloc(576 * 128 * sizeof(_Float16));
    _Float16* Wg2_p = (_Float16*)alloc(128 * 16 * sizeof(_Float16));

    hipMemsetAsync(agg_s, 0, (size_t)N * DS_ * sizeof(float), stream);
    hipMemsetAsync(agg_V, 0, (size_t)N * 48 * sizeof(float), stream);

    pack_w<<<160, 256, 0, stream>>>(Ws_m, Wsm_p, 320, 128);
    pack_w<<<8,   256, 0, stream>>>((const float*)d_in[8], Wgm_p, 128, 16);
    pack_w<<<256, 256, 0, stream>>>(Ws1, Ws1_p, 384, 512);
    pack_w<<<64,  256, 0, stream>>>(Wg1, Wg1_p, 512, 32);
    pack_w<<<160, 256, 0, stream>>>(Ws2, Ws2_p, 576, 128);
    pack_w<<<8,   256, 0, stream>>>(Wg2, Wg2_p, 128, 16);

    edge_kernel<<<E / 32, 64, 0, stream>>>(
        node_s, node_V, frames, eidx, Wvd_m, Wvu_m, bs_m, bg_m, Watt, batt,
        Wsm_p, Wgm_p, agg_s, agg_V, E);

    float* out_s = (float*)d_out;
    float* out_V = out_s + (size_t)N * DS_;
    node_kernel<<<N / 16, 32, 0, stream>>>(
        node_s, node_V, frames, agg_s, agg_V,
        bs1, bg1, bs2, bg2, Wvd1, Wvu1, Wvd2, Wvu2,
        Ws1_p, Wg1_p, Ws2_p, Wg2_p, gamma, beta, out_s, out_V);
}
